// ResidualVQ_79963701117574
// MI455X (gfx1250) — matrix operand order repaired and checked by reading, not compiled, not run on hardware
//
#include <hip/hip_runtime.h>
#include <hip/hip_bf16.h>

// ---------------------------------------------------------------------------
// ResidualVQ (eval forward), 6 layers, B=64 C=512 T=256 K=1024.
// Distance GEMM on v_wmma_f32_16x16x32_bf16:
//   - 64-row M-tiling (4x B reuse per fragment)
//   - codebook pre-swizzled into WMMA fragment layout (coalesced 1KB frags)
//   - A kept in LDS (hoist blocked) so B can pipeline + occupancy stays high
// ---------------------------------------------------------------------------

typedef __attribute__((ext_vector_type(16))) __bf16 v16bf;
typedef __attribute__((ext_vector_type(8)))  float  v8f;

#define B_ 64
#define C_ 512
#define T_ 256
#define Q_ 6
#define K_ 1024
#define N_ (B_ * T_)            // 16384 rows of xf
#define QOUT_N (B_ * C_ * T_)   // 8388608
#define IDXOUT_N (N_ * Q_)      // 98304
#define LDSROW (C_ + 8)         // pad LDS rows (16B) to break bank conflicts
#define MT_ 4                   // M-tiles per block
#define ROWS_ (MT_ * 16)        // 64 rows per block
#define NT_ (K_ / 16)           // 64 N-tiles per codebook
#define KT_ (C_ / 32)           // 16 k-steps
#define FRAG_ 512               // elements per B fragment (32 lanes x 16 bf16)

__device__ __forceinline__ unsigned short f2bf(float f) {
    unsigned u = __float_as_uint(f);
    u += 0x7FFFu + ((u >> 16) & 1u);     // round-to-nearest-even
    return (unsigned short)(u >> 16);
}

// ---------------- init: residual = x, qout = 0 ----------------
__global__ void init_main(const float* __restrict__ x,
                          float* __restrict__ res,
                          float* __restrict__ qout) {
    int i = blockIdx.x * blockDim.x + threadIdx.x;
    if (i < QOUT_N) { res[i] = x[i]; qout[i] = 0.0f; }
}

__global__ void init_small(unsigned* __restrict__ counts,
                           float* __restrict__ commitAcc) {
    int i = blockIdx.x * blockDim.x + threadIdx.x;
    if (i < Q_ * K_) counts[i] = 0u;
    if (i < Q_) commitAcc[i] = 0.0f;
}

// ---- codebook f32 -> bf16 in WMMA B-fragment layout + ||cb||^2 ----
// Fragment layout: cbhF[((q*NT_ + ntile)*KT_ + kt)*32 + hi*16 + m][16]
// so one fragment = 1KB contiguous; lane reads 32B at lane*32. Uses the
// contiguous k order of the A loads (lanes 0-15: K 0..15, lanes 16-31: K 16..31).
__global__ __launch_bounds__(128) void cvt_codebook(const float* __restrict__ cb,
                                                    unsigned short* __restrict__ cbhF,
                                                    float* __restrict__ cbnorm) {
    __shared__ float red[128];
    const int code = blockIdx.x;                    // q*K + k
    const int q  = code >> 10;
    const int kl = code & (K_ - 1);
    const int ntile = kl >> 4;
    const int m     = kl & 15;
    const size_t base = (size_t)code * C_;
    float s = 0.0f;
    #pragma unroll
    for (int j = 0; j < C_ / 128; ++j) {
        int c = j * 128 + threadIdx.x;
        float v = cb[base + c];
        const int kt = c >> 5;
        const int hi = (c >> 4) & 1;
        const int jj = c & 15;
        size_t dst = ((((size_t)(q * NT_ + ntile) * KT_ + kt) * 32) + hi * 16 + m) * 16 + jj;
        cbhF[dst] = f2bf(v);
        s += v * v;
    }
    red[threadIdx.x] = s;
    __syncthreads();
    for (int o = 64; o > 0; o >>= 1) {
        if (threadIdx.x < o) red[threadIdx.x] += red[threadIdx.x + o];
        __syncthreads();
    }
    if (threadIdx.x == 0) cbnorm[code] = red[0];
}

// ---------------- WMMA distance + argmin ----------------
// 1 block = 64 rows of xf (4 M-tiles); 4 waves each sweep 16 of 64 N-tiles.
__global__ __launch_bounds__(128) void argmin_wmma(
    const float* __restrict__ res, const unsigned short* __restrict__ cbhF,
    const float* __restrict__ cbnorm, int* __restrict__ idxWs,
    float* __restrict__ idxOut, int q) {
    __shared__ unsigned short ldsX[ROWS_ * LDSROW];     // 64x520 bf16 = 65 KB
    __shared__ float sdist[4][ROWS_];
    __shared__ int   sidx[4][ROWS_];

    const int tid = threadIdx.x;
    const int rowBase = blockIdx.x * ROWS_;         // 64 rows share one b (T=256)
    const int b = rowBase >> 8;
    const int tBase = rowBase & 255;

    // Stage residual tile (64 rows x 512 C) as bf16 in LDS, t-coalesced.
    for (int e = tid; e < ROWS_ * C_; e += 128) {
        int tl = e & (ROWS_ - 1);
        int c  = e >> 6;
        float v = res[(size_t)b * (C_ * T_) + (size_t)c * T_ + (tBase + tl)];
        ldsX[tl * LDSROW + c] = f2bf(v);
    }
    __syncthreads();

    const int wave = tid >> 5;
    const int lane = tid & 31;
    const int m    = lane & 15;
    const int hi   = (lane >> 4) & 1;

    float bestD[MT_][8];
    int   bestI[MT_][8];
    #pragma unroll
    for (int mt = 0; mt < MT_; ++mt)
        #pragma unroll
        for (int r = 0; r < 8; ++r) { bestD[mt][r] = 3.402823466e+38f; bestI[mt][r] = 0; }

    const float* cnq = cbnorm + q * K_;
    // Per-lane fragment pointer: frag(ct,kt) = base + (ct*KT_ + kt)*FRAG_ + lane*16
    const unsigned short* fp0 =
        cbhF + (size_t)(q * NT_) * KT_ * FRAG_ + (size_t)lane * 16;

    for (int ct = wave; ct < NT_; ct += 4) {
        const int n = ct * 16 + m;
        const unsigned short* fp = fp0 + (size_t)ct * KT_ * FRAG_;
        // Prefetch this wave's next ct iteration (pure sequential +64KB stream).
        __builtin_prefetch((const void*)(fp + (size_t)4 * KT_ * FRAG_), 0, 1);

        // Opaque zero defined inside the loop: keeps A-fragment ds_loads in the
        // loop body (blocks LICM from hoisting 512 VGPRs of A and starving the
        // B pipeline / occupancy).
        int zofs = 0;
        asm volatile("" : "+v"(zofs));
        const unsigned short* ap[MT_];
        #pragma unroll
        for (int mt = 0; mt < MT_; ++mt)
            ap[mt] = ldsX + (mt * 16 + m) * LDSROW + hi * 16 + zofs;

        v8f acc[MT_];
        #pragma unroll
        for (int mt = 0; mt < MT_; ++mt) acc[mt] = (v8f){};

        #pragma unroll
        for (int ktg = 0; ktg < KT_ / 4; ++ktg) {
            // Load 4 B fragments (8 b128 in flight) before consuming them.
            v16bf Bf[4];
            #pragma unroll
            for (int j = 0; j < 4; ++j)
                Bf[j] = *(const v16bf*)(fp + (size_t)(ktg * 4 + j) * FRAG_);
            #pragma unroll
            for (int j = 0; j < 4; ++j) {
                const int c0 = (ktg * 4 + j) * 32;
                #pragma unroll
                for (int mt = 0; mt < MT_; ++mt) {
                    // A (16x32 bf16): lane<16 K {0..15}, lane>=16 {16..31}
                    union { uint4 u[2]; v16bf v; } A;
                    A.u[0] = *(const uint4*)(ap[mt] + c0);
                    A.u[1] = *(const uint4*)(ap[mt] + c0 + 8);
                    acc[mt] = __builtin_amdgcn_wmma_f32_16x16x32_bf16(
                        false, A.v, false, Bf[j], (short)0, acc[mt], false, false);
                }
            }
        }
        const float cn = cnq[n];
        #pragma unroll
        for (int mt = 0; mt < MT_; ++mt) {
            #pragma unroll
            for (int r = 0; r < 8; ++r) {
                float d = cn - 2.0f * acc[mt][r];  // ||cb||^2 - 2 x.cb
                if (d < bestD[mt][r] || (d == bestD[mt][r] && n < bestI[mt][r])) {
                    bestD[mt][r] = d; bestI[mt][r] = n;
                }
            }
        }
    }

    // Butterfly min across the 16 lanes holding the 16 N values of each row.
    #pragma unroll
    for (int mt = 0; mt < MT_; ++mt) {
        #pragma unroll
        for (int r = 0; r < 8; ++r) {
            #pragma unroll
            for (int off = 8; off >= 1; off >>= 1) {
                float od = __shfl_xor(bestD[mt][r], off, 16);
                int   oi = __shfl_xor(bestI[mt][r], off, 16);
                if (od < bestD[mt][r] || (od == bestD[mt][r] && oi < bestI[mt][r])) {
                    bestD[mt][r] = od; bestI[mt][r] = oi;
                }
            }
        }
    }
    if (m == 0) {   // lane 0 -> rows hi*8 + 0..7 of each M-tile (hi=0/1)
        #pragma unroll
        for (int mt = 0; mt < MT_; ++mt) {
            #pragma unroll
            for (int r = 0; r < 8; ++r) {
                sdist[wave][mt * 16 + hi * 8 + r] = bestD[mt][r];
                sidx [wave][mt * 16 + hi * 8 + r] = bestI[mt][r];
            }
        }
    }
    __syncthreads();
    if (tid < ROWS_) {
        float bd = sdist[0][tid]; int bi = sidx[0][tid];
        #pragma unroll
        for (int w = 1; w < 4; ++w) {
            float od = sdist[w][tid]; int oi = sidx[w][tid];
            if (od < bd || (od == bd && oi < bi)) { bd = od; bi = oi; }
        }
        const int i = rowBase + tid;
        idxWs[i] = bi;
        idxOut[(size_t)i * Q_ + q] = (float)bi;  // all_indices is (B,T,Q)
    }
}

// ---------------- dequantize, residual update, commit loss ----------------
__global__ __launch_bounds__(256) void dequant_update(
    const float* __restrict__ cbq, const int* __restrict__ idxWs,
    float* __restrict__ res, float* __restrict__ qout,
    float* __restrict__ commitAcc, int q) {
    __shared__ float red[256];
    const int c = blockIdx.x;
    const int b = blockIdx.y;
    const int t = threadIdx.x;
    const int i = b * T_ + t;
    const int id = idxWs[i];
    const float xd = cbq[(size_t)id * C_ + c];
    const size_t off = ((size_t)b * C_ + c) * T_ + t;
    const float diff = res[off] - xd;     // commit uses residual BEFORE update
    res[off] = diff;
    qout[off] += xd;
    red[t] = diff * diff;
    __syncthreads();
    for (int o = 128; o > 0; o >>= 1) {
        if (t < o) red[t] += red[t + o];
        __syncthreads();
    }
    if (t == 0) atomicAdd(commitAcc + q, red[0]);
}

// ---------------- code-usage histogram ----------------
__global__ void histo(const int* __restrict__ idxWs,
                      unsigned* __restrict__ counts, int q) {
    int i = blockIdx.x * blockDim.x + threadIdx.x;
    if (i < N_) atomicAdd(&counts[q * K_ + idxWs[i]], 1u);
}

// ---------------- perplexity + mean commit ----------------
__global__ __launch_bounds__(256) void finalize_stats(
    const unsigned* __restrict__ counts, const float* __restrict__ commitAcc,
    float* __restrict__ out) {
    __shared__ float red[256];
    __shared__ float perpSum;
    if (threadIdx.x == 0) perpSum = 0.0f;
    __syncthreads();
    for (int q = 0; q < Q_; ++q) {
        float h = 0.0f;
        for (int k = threadIdx.x; k < K_; k += 256) {
            float p = (float)counts[q * K_ + k] * (1.0f / (float)N_);
            h -= p * __logf(p + 1e-7f);
        }
        red[threadIdx.x] = h;
        __syncthreads();
        for (int o = 128; o > 0; o >>= 1) {
            if (threadIdx.x < o) red[threadIdx.x] += red[threadIdx.x + o];
            __syncthreads();
        }
        if (threadIdx.x == 0) perpSum += __expf(red[0]);
        __syncthreads();
    }
    if (threadIdx.x == 0) {
        float cs = 0.0f;
        for (int q = 0; q < Q_; ++q) cs += commitAcc[q];
        out[QOUT_N + IDXOUT_N + 0] = cs / ((float)Q_ * (float)N_ * (float)C_);
        out[QOUT_N + IDXOUT_N + 1] = perpSum / (float)Q_;
    }
}

// ---------------------------------------------------------------------------
extern "C" void kernel_launch(void* const* d_in, const int* in_sizes, int n_in,
                              void* d_out, int out_size, void* d_ws, size_t ws_size,
                              hipStream_t stream) {
    const float* x  = (const float*)d_in[0];   // (B,C,T)
    const float* cb = (const float*)d_in[1];   // (Q,K,C)
    float* out    = (float*)d_out;
    float* qout   = out;                       // (B,C,T)
    float* idxOut = out + QOUT_N;              // (B,T,Q) as float

    char* ws = (char*)d_ws;
    size_t o = 0;
    auto alloc = [&](size_t bytes) -> void* {
        void* p = ws + o;
        o = (o + bytes + 255) & ~(size_t)255;
        return p;
    };
    float*          res     = (float*)          alloc((size_t)QOUT_N * 4);        // 33.5 MB
    unsigned short* cbhF    = (unsigned short*) alloc((size_t)Q_ * K_ * C_ * 2);  //  6.3 MB
    float*          cbnorm  = (float*)          alloc((size_t)Q_ * K_ * 4);
    int*            idxWs   = (int*)            alloc((size_t)N_ * 4);
    unsigned*       counts  = (unsigned*)       alloc((size_t)Q_ * K_ * 4);
    float*          commitA = (float*)          alloc(256);
    (void)in_sizes; (void)n_in; (void)out_size; (void)ws_size;

    init_main <<<(QOUT_N + 255) / 256, 256, 0, stream>>>(x, res, qout);
    init_small<<<(Q_ * K_ + 255) / 256, 256, 0, stream>>>(counts, commitA);
    cvt_codebook<<<Q_ * K_, 128, 0, stream>>>(cb, cbhF, cbnorm);

    for (int q = 0; q < Q_; ++q) {
        argmin_wmma<<<N_ / ROWS_, 128, 0, stream>>>(res, cbhF, cbnorm, idxWs, idxOut, q);
        dequant_update<<<dim3(C_, B_), 256, 0, stream>>>(cb + (size_t)q * K_ * C_,
                                                         idxWs, res, qout, commitA, q);
        histo<<<N_ / 256, 256, 0, stream>>>(idxWs, counts, q);
    }
    finalize_stats<<<1, 256, 0, stream>>>(counts, commitA, out);
}
